// detLSTM_stochastic_RNN_autoreg_torchscript_38293928411399
// MI455X (gfx1250) — matrix operand order repaired and checked by reading, not compiled, not run on hardware
//
#include <hip/hip_runtime.h>
#include <hip/hip_bf16.h>
#include <math.h>

typedef __attribute__((ext_vector_type(16))) __bf16       v16bf;
typedef __attribute__((ext_vector_type(8)))  float        v8f;
typedef __attribute__((ext_vector_type(8)))  unsigned int v8u;

#define B_TOT   2048
#define LSEQ    60
#define NX      9
#define NH_MEM  128
#define H       256
#define NXS     17
#define NY      5
#define NYS     8
#define K1P     416      // 160 (x padded 137->160) + 256 (h)
#define K2P     512      // 256 (h1 in) + 256 (h2)
#define KA      520      // LDS A-matrix row stride (bf16 elems), conflict-skewed
#define SSTR    264      // fp32 staging row stride

// workspace byte offsets (all 64B aligned)
#define WS_W1   0
#define WS_W2   (WS_W1 + 1024*K1P*2)          // 851968
#define WS_WLAT (WS_W2 + 1024*K2P*2)          // 1900544
#define WS_H1   (WS_WLAT + 128*H*2)           // 1966080  (+62.9MB h1 sequence)

__device__ __forceinline__ unsigned short f2bfb(float f) {
  unsigned u = __builtin_bit_cast(unsigned, f);
  u += 0x7FFFu + ((u >> 16) & 1u);             // round-to-nearest-even
  return (unsigned short)(u >> 16);
}
__device__ __forceinline__ float bfb2f(unsigned short s) {
  unsigned u = ((unsigned)s) << 16;
  return __builtin_bit_cast(float, u);
}
__device__ __forceinline__ float sigm(float x) { return 1.0f / (1.0f + __expf(-x)); }

__device__ __forceinline__ v8f splat8(float x) {
  v8f v;
#pragma unroll
  for (int e = 0; e < 8; ++e) v[e] = x;
  return v;
}

// A fragment: 16x32 bf16 tile from LDS, contiguous K order (k = 16*hi + s), same as load_b.
// lanes 0-15 = rows 0-15 (K 0..15); lanes 16-31 = rows 0-15 (K 16..31)
__device__ __forceinline__ v16bf load_a(const unsigned short* sArow, int k0, int hi) {
  v8u u;
  const int kb = k0 + hi * 16;
#pragma unroll
  for (int p = 0; p < 8; ++p) {
    int k = kb + 2 * p;
    u[p] = *(const unsigned int*)(sArow + k);
  }
  return __builtin_bit_cast(v16bf, u);
}

// B fragment: 32x16 (KxN) tile, element(k,n)=W[n,k] from row-major packed bf16.
// Per-lane data is 32 contiguous bytes -> two b128 loads.
__device__ __forceinline__ v16bf load_b(const unsigned short* wrow, int k0) {
  const uint4* p = (const uint4*)(wrow + k0);
  uint4 lo = p[0], hh = p[1];
  v8u u;
  u[0] = lo.x; u[1] = lo.y; u[2] = lo.z; u[3] = lo.w;
  u[4] = hh.x; u[5] = hh.y; u[6] = hh.z; u[7] = hh.w;
  return __builtin_bit_cast(v16bf, u);
}

__device__ __forceinline__ v8f wmma_bf16(v16bf a, v16bf b, v8f c) {
  return __builtin_amdgcn_wmma_f32_16x16x32_bf16(false, a, false, b, (short)0, c,
                                                 false, false);
}

// ---------------------------------------------------------------------------
// Kernel 1: pack fp32 weights into WMMA-friendly bf16 layouts in workspace.
//   W1 = [Wih1 (137, zero-padded to 160) | Whh1 (256)]  -> 1024 x 416
//   W2 = [Wih2 (256) | Whh2 (256)]                      -> 1024 x 512
//   WL = Wlat                                           ->  128 x 256
// ---------------------------------------------------------------------------
__global__ void pack_weights(const float* __restrict__ Wih1, const float* __restrict__ Whh1,
                             const float* __restrict__ Wih2, const float* __restrict__ Whh2,
                             const float* __restrict__ Wlat, unsigned short* __restrict__ ws) {
  unsigned short* W1 = ws + WS_W1 / 2;
  unsigned short* W2 = ws + WS_W2 / 2;
  unsigned short* WL = ws + WS_WLAT / 2;
  const int T1 = 1024 * K1P, T2 = 1024 * K2P, T3 = 128 * H;
  const int total = T1 + T2 + T3;
  for (int i = blockIdx.x * blockDim.x + threadIdx.x; i < total;
       i += gridDim.x * blockDim.x) {
    if (i < T1) {
      int n = i / K1P, k = i % K1P;
      float v = (k < 160) ? ((k < 137) ? Wih1[n * 137 + k] : 0.0f)
                          : Whh1[n * H + (k - 160)];
      W1[i] = f2bfb(v);
    } else if (i < T1 + T2) {
      int j = i - T1, n = j / K2P, k = j % K2P;
      float v = (k < H) ? Wih2[n * H + k] : Whh2[n * H + (k - H)];
      W2[j] = f2bfb(v);
    } else {
      int j = i - T1 - T2;
      WL[j] = f2bfb(Wlat[j]);
    }
  }
}

// ---------------------------------------------------------------------------
// Kernel 2: persistent fused 2-layer LSTM + heads. One block per 16 batch rows.
// 256 threads = 8 wave32; wave w owns units [w*32, w*32+32) across all 4 gates.
// ---------------------------------------------------------------------------
__global__ __launch_bounds__(256)
void rnn_persistent(const float* __restrict__ inputs_main, const float* __restrict__ inputs_aux,
                    const float* __restrict__ rnn1_mem,
                    const float* __restrict__ Ws,  const float* __restrict__ bs,
                    const float* __restrict__ Ws2, const float* __restrict__ bs2,
                    const float* __restrict__ Wt,  const float* __restrict__ bt,
                    const float* __restrict__ Wt2, const float* __restrict__ bt2,
                    const float* __restrict__ b1,  const float* __restrict__ b2,
                    const float* __restrict__ blat,
                    const float* __restrict__ Wout, const float* __restrict__ bout,
                    const float* __restrict__ Wsfc, const float* __restrict__ bsfc,
                    unsigned short* __restrict__ wsw,
                    float* __restrict__ out, float* __restrict__ out_sfc,
                    float* __restrict__ mem_new) {
  __shared__ unsigned short sA[16 * KA];   // bf16 A operand: [x|h1] or [h1in|h2]
  __shared__ float sStage[16 * SSTR];      // fp32 staging (c init, lat)

  const int tid  = threadIdx.x;
  const int wave = tid >> 5;
  const int lane = tid & 31;
  const int lrow = lane & 15;   // A row / C-D column within tile
  const int hi   = lane >> 4;
  const int m0   = blockIdx.x * 16;

  const unsigned short* W1 = wsw + WS_W1 / 2;
  const unsigned short* W2 = wsw + WS_W2 / 2;
  const unsigned short* WL = wsw + WS_WLAT / 2;
  unsigned short*       H1 = wsw + WS_H1 / 2;   // bf16 h1 sequence [b][t][u]

  // per-wave weight row pointers + biases: gate g, half-tile t -> 16 columns
  const unsigned short* bptr1[4][2];
  const unsigned short* bptr2[4][2];
  float bias1[4][2], bias2[4][2];
#pragma unroll
  for (int g = 0; g < 4; ++g)
#pragma unroll
    for (int t = 0; t < 2; ++t) {
      int n = g * H + wave * 32 + t * 16 + lrow;
      bptr1[g][t] = W1 + (size_t)n * K1P + hi * 16;
      bptr2[g][t] = W2 + (size_t)n * K2P + hi * 16;
      bias1[g][t] = b1[n];
      bias2[g][t] = b2[n];
    }
  const unsigned short* sArow = sA + lrow * KA;
  float cst[2][8];

  // ---------------- layer 1 init: h=tanh(aux@Ws.T+bs), c=tanh(aux@Ws2.T+bs2)
  for (int idx = tid; idx < 16 * H; idx += 256) {
    int r = idx >> 8, u = idx & 255;
    const float* a = inputs_aux + (size_t)(m0 + r) * NXS;
    float hv = bs[u], cv = bs2[u];
#pragma unroll
    for (int k = 0; k < NXS; ++k) { float av = a[k]; hv += av * Ws[u * NXS + k]; cv += av * Ws2[u * NXS + k]; }
    sA[r * KA + 160 + u] = f2bfb(tanhf(hv));
    sStage[r * SSTR + u] = tanhf(cv);
  }
  __syncthreads();
#pragma unroll
  for (int t = 0; t < 2; ++t) {
    int u = wave * 32 + t * 16 + lrow;
#pragma unroll
    for (int r = 0; r < 8; ++r) cst[t][r] = sStage[(r + hi * 8) * SSTR + u];
  }
  __syncthreads();

  // ---------------- layer 1: flipped sweep, lev = 59-l ----------------------
  for (int l = 0; l < LSEQ; ++l) {
    const int lev = LSEQ - 1 - l;
    for (int idx = tid; idx < 16 * 160; idx += 256) {     // stage x tile (bf16)
      int r = idx / 160, c = idx % 160;
      float v = 0.0f;
      if (c < NX)        v = inputs_main[((size_t)(m0 + r) * LSEQ + lev) * NX + c];
      else if (c < 137)  v = rnn1_mem[((size_t)(m0 + r) * LSEQ + lev) * NH_MEM + (c - NX)];
      sA[r * KA + c] = f2bfb(v);
    }
    __syncthreads();

    v8f acc[4][2];
#pragma unroll
    for (int g = 0; g < 4; ++g)
#pragma unroll
      for (int t = 0; t < 2; ++t) acc[g][t] = splat8(bias1[g][t]);

    for (int k0 = 0; k0 < K1P; k0 += 32) {                // 13 K-steps, 8 WMMA each
      v16bf a = load_a(sArow, k0, hi);
#pragma unroll
      for (int g = 0; g < 4; ++g)
#pragma unroll
        for (int t = 0; t < 2; ++t)
          acc[g][t] = wmma_bf16(a, load_b(bptr1[g][t], k0), acc[g][t]);
    }
    __syncthreads();

#pragma unroll
    for (int t = 0; t < 2; ++t) {                         // in-wave cell update
      int u = wave * 32 + t * 16 + lrow;
#pragma unroll
      for (int r = 0; r < 8; ++r) {
        int m = r + hi * 8;
        float c = sigm(acc[1][t][r]) * cst[t][r] + sigm(acc[0][t][r]) * tanhf(acc[2][t][r]);
        float hv = sigm(acc[3][t][r]) * tanhf(c);
        cst[t][r] = c;
        unsigned short hb = f2bfb(hv);
        sA[m * KA + 160 + u] = hb;                                       // next-step h
        H1[((size_t)(m0 + m) * LSEQ + lev) * H + u] = hb;                // rnn1out[lev]
      }
    }
  }
  __threadfence_block();

  // ---------------- layer 2 init from toa=[aux1,aux6] -----------------------
  __syncthreads();
  for (int idx = tid; idx < 16 * H; idx += 256) {
    int r = idx >> 8, u = idx & 255;
    float t0 = inputs_aux[(size_t)(m0 + r) * NXS + 1];
    float t1 = inputs_aux[(size_t)(m0 + r) * NXS + 6];
    float hv = bt[u]  + t0 * Wt[u * 2]  + t1 * Wt[u * 2 + 1];
    float cv = bt2[u] + t0 * Wt2[u * 2] + t1 * Wt2[u * 2 + 1];
    sA[r * KA + 256 + u] = f2bfb(tanhf(hv));
    sStage[r * SSTR + u] = tanhf(cv);
  }
  __syncthreads();
#pragma unroll
  for (int t = 0; t < 2; ++t) {
    int u = wave * 32 + t * 16 + lrow;
#pragma unroll
    for (int r = 0; r < 8; ++r) cst[t][r] = sStage[(r + hi * 8) * SSTR + u];
  }
  __syncthreads();

  const float biasl = blat[wave * 16 + lrow];
  const unsigned short* bptrL = WL + (size_t)(wave * 16 + lrow) * H + hi * 16;

  // ---------------- layer 2 forward sweep + fused heads ---------------------
  for (int t = 0; t < LSEQ; ++t) {
    for (int idx = tid; idx < 16 * 128; idx += 256) {     // stage h1in (bf16 copy)
      int r = idx >> 7, cu = idx & 127;
      const unsigned int* src = (const unsigned int*)(H1 + ((size_t)(m0 + r) * LSEQ + t) * H);
      *(unsigned int*)(sA + r * KA + 2 * cu) = src[cu];
    }
    __syncthreads();

    v8f acc[4][2];
#pragma unroll
    for (int g = 0; g < 4; ++g)
#pragma unroll
      for (int tt = 0; tt < 2; ++tt) acc[g][tt] = splat8(bias2[g][tt]);

    for (int k0 = 0; k0 < K2P; k0 += 32) {                // 16 K-steps, 8 WMMA each
      v16bf a = load_a(sArow, k0, hi);
#pragma unroll
      for (int g = 0; g < 4; ++g)
#pragma unroll
        for (int tt = 0; tt < 2; ++tt)
          acc[g][tt] = wmma_bf16(a, load_b(bptr2[g][tt], k0), acc[g][tt]);
    }
    __syncthreads();

#pragma unroll
    for (int tt = 0; tt < 2; ++tt) {
      int u = wave * 32 + tt * 16 + lrow;
#pragma unroll
      for (int r = 0; r < 8; ++r) {
        int m = r + hi * 8;
        float c = sigm(acc[1][tt][r]) * cst[tt][r] + sigm(acc[0][tt][r]) * tanhf(acc[2][tt][r]);
        float hv = sigm(acc[3][tt][r]) * tanhf(c);
        cst[tt][r] = c;
        sA[m * KA + 256 + u] = f2bfb(hv);
      }
    }
    __syncthreads();

    // lat = h2 @ Wlat.T + blat  (16x256x128, one N-tile per wave)
    v8f la = splat8(biasl);
    for (int k0 = 0; k0 < H; k0 += 32)
      la = wmma_bf16(load_a(sArow, 256 + k0, hi), load_b(bptrL, k0), la);

    const int n = wave * 16 + lrow;
#pragma unroll
    for (int r = 0; r < 8; ++r) {
      int m = r + hi * 8;
      mem_new[((size_t)(m0 + m) * LSEQ + (LSEQ - 1 - t)) * NH_MEM + n] = la[r];
      sStage[m * SSTR + n] = la[r];
    }
    __syncthreads();

    if (tid < 16 * NY) {                                  // out = lat @ Wout.T + bout
      int r = tid / NY, j = tid % NY;
      float s = bout[j];
      for (int k = 0; k < NH_MEM; ++k) s += sStage[r * SSTR + k] * Wout[j * NH_MEM + k];
      out[((size_t)(m0 + r) * LSEQ + t) * NY + j] = s;
    }
    __syncthreads();
  }

  // out_sfc = relu(last_hidden @ Wsfc.T + bsfc), h2 still in sA[.,256..511]
  if (tid < 16 * NYS) {
    int r = tid >> 3, j = tid & 7;
    float s = bsfc[j];
    for (int k = 0; k < H; ++k) s += bfb2f(sA[r * KA + 256 + k]) * Wsfc[j * H + k];
    out_sfc[(size_t)(m0 + r) * NYS + j] = fmaxf(s, 0.0f);
  }
}

extern "C" void kernel_launch(void* const* d_in, const int* in_sizes, int n_in,
                              void* d_out, int out_size, void* d_ws, size_t ws_size,
                              hipStream_t stream) {
  (void)in_sizes; (void)n_in; (void)out_size; (void)ws_size;
  const float* inputs_main = (const float*)d_in[0];
  const float* inputs_aux  = (const float*)d_in[1];
  const float* rnn1_mem    = (const float*)d_in[2];
  const float* Ws   = (const float*)d_in[3];
  const float* bs   = (const float*)d_in[4];
  const float* Ws2  = (const float*)d_in[5];
  const float* bs2  = (const float*)d_in[6];
  const float* Wt   = (const float*)d_in[7];
  const float* bt   = (const float*)d_in[8];
  const float* Wt2  = (const float*)d_in[9];
  const float* bt2  = (const float*)d_in[10];
  const float* Wih1 = (const float*)d_in[11];
  const float* Whh1 = (const float*)d_in[12];
  const float* b1   = (const float*)d_in[13];
  const float* Wih2 = (const float*)d_in[14];
  const float* Whh2 = (const float*)d_in[15];
  const float* b2   = (const float*)d_in[16];
  const float* Wlat = (const float*)d_in[17];
  const float* blat = (const float*)d_in[18];
  const float* Wout = (const float*)d_in[19];
  const float* bout = (const float*)d_in[20];
  const float* Wsfc = (const float*)d_in[21];
  const float* bsfc = (const float*)d_in[22];

  unsigned short* wsw = (unsigned short*)d_ws;
  float* out     = (float*)d_out;
  float* out_sfc = out + (size_t)B_TOT * LSEQ * NY;
  float* mem_new = out_sfc + (size_t)B_TOT * NYS;

  pack_weights<<<1024, 256, 0, stream>>>(Wih1, Whh1, Wih2, Whh2, Wlat, wsw);
  rnn_persistent<<<B_TOT / 16, 256, 0, stream>>>(
      inputs_main, inputs_aux, rnn1_mem, Ws, bs, Ws2, bs2, Wt, bt, Wt2, bt2,
      b1, b2, blat, Wout, bout, Wsfc, bsfc, wsw, out, out_sfc, mem_new);
}
